// sca_49194555408679
// MI455X (gfx1250) — compile-verified, matrix operand order checked by reading, not run on hardware
//
#include <hip/hip_runtime.h>

#define NHD 8    // num heads
#define NPT 8    // num sampling points
#define EDIM 256
#define DH 32    // channels per head

typedef __attribute__((ext_vector_type(16))) _Float16 v16h;
typedef __attribute__((ext_vector_type(8)))  float    v8f;
typedef __attribute__((ext_vector_type(4)))  float    f32x4;

// ---------------------------------------------------------------------------
// Kernel 1: per-batch projections via WMMA (one workgroup, 12 waves).
//   proj[bs x 192] = relu( q @ [W_off | W_attn] + [b_off | b_attn] )
//   cols 0..127  -> sampling offsets, normalized by (w,h)
//   cols 128..191-> attention logits, softmax over NP per (b,head)
// ---------------------------------------------------------------------------
__global__ void __launch_bounds__(384)
proj_wmma_kernel(const float* __restrict__ query,    // [bs,1,ED]
                 const float* __restrict__ W_off,    // [ED, NH*NP*2]
                 const float* __restrict__ b_off,    // [NH*NP*2]
                 const float* __restrict__ W_attn,   // [ED, NH*NP]
                 const float* __restrict__ b_attn,   // [NH*NP]
                 const int*   __restrict__ hptr,
                 const int*   __restrict__ wptr,
                 float* __restrict__ off_ws,         // [bs,NH,NP,2] normalized
                 float* __restrict__ aw_ws,          // [bs,NH,NP] softmaxed
                 int bs)
{
    __shared__ float proj[8][192];

    const int lane = threadIdx.x & 31;
    const int tile = threadIdx.x >> 5;           // 0..11 : one 16-col N tile per wave
    const int row  = lane & 15;                  // M index (batch row, padded to 16)
    const int khi  = (lane >> 4) * 8;            // +8 K offset for lanes 16..31
    const int colg = tile * 16 + (lane & 15);    // global output column 0..191

    // Unconditional, in-bounds A-row pointer; padding rows zeroed by select.
    const float* qrow = query + min(row, bs - 1) * EDIM;
    const bool   arow_ok = (row < bs);

    // Resolve [W_off | W_attn] split once per wave (colg<128 is tile-uniform).
    const bool   isOff   = (colg < 128);
    const float* Wcol    = isOff ? (W_off + colg) : (W_attn + (colg - 128));
    const int    wstride = isOff ? 128 : 64;

    v8f acc = {};
    #pragma unroll
    for (int kk = 0; kk < EDIM; kk += 32) {
        // A fragment: e<8 -> K = kk+khi+e ; e>=8 -> K = kk+16+khi+(e&7)
        f32x4 qa = *(const f32x4*)(qrow + kk + khi);
        f32x4 qb = *(const f32x4*)(qrow + kk + khi + 4);
        f32x4 qc = *(const f32x4*)(qrow + kk + 16 + khi);
        f32x4 qd = *(const f32x4*)(qrow + kk + 16 + khi + 4);

        v16h afrag, bfrag;
        #pragma unroll
        for (int j = 0; j < 4; ++j) {
            afrag[j]      = (_Float16)(arow_ok ? qa[j] : 0.0f);
            afrag[4 + j]  = (_Float16)(arow_ok ? qb[j] : 0.0f);
            afrag[8 + j]  = (_Float16)(arow_ok ? qc[j] : 0.0f);
            afrag[12 + j] = (_Float16)(arow_ok ? qd[j] : 0.0f);
        }
        #pragma unroll
        for (int e = 0; e < 16; ++e) {
            int k = kk + ((e >> 3) << 4) + khi + (e & 7);
            bfrag[e] = (_Float16)Wcol[k * wstride];
        }
        acc = __builtin_amdgcn_wmma_f32_16x16x32_f16(
            /*neg_a=*/false, afrag, /*neg_b=*/false, bfrag,
            /*c_mod=*/(short)0, acc, /*reuse_a=*/false, /*reuse_b=*/false);
    }

    // D layout: lanes 0-15 hold M=0..7 across VGPR 0..7 (exactly our bs rows).
    if (lane < 16) {
        float bias = isOff ? b_off[colg] : b_attn[colg - 128];
        #pragma unroll
        for (int r = 0; r < 8; ++r) {
            if (r < bs) {
                float v = acc[r] + bias;
                proj[r][colg] = v > 0.0f ? v : 0.0f;   // relu
            }
        }
    }
    __syncthreads();

    const int H = *hptr, W = *wptr;
    const int tid = threadIdx.x;

    // softmax over points (attention part, cols 128..191)
    if (tid < bs * NHD) {
        int b = tid >> 3, hh = tid & 7;
        float v[NPT], m = -1e30f;
        #pragma unroll
        for (int p = 0; p < NPT; ++p) {
            v[p] = proj[b][128 + hh * NPT + p];
            m = fmaxf(m, v[p]);
        }
        float s = 0.0f;
        #pragma unroll
        for (int p = 0; p < NPT; ++p) { v[p] = __expf(v[p] - m); s += v[p]; }
        float inv = 1.0f / s;
        #pragma unroll
        for (int p = 0; p < NPT; ++p) aw_ws[(b * NHD + hh) * NPT + p] = v[p] * inv;
    }

    // normalized offsets (cols 0..127): x / W, y / H
    for (int i = tid; i < bs * 128; i += (int)blockDim.x) {
        int b = i >> 7, c = i & 127;
        float norm = (c & 1) ? (float)H : (float)W;
        off_ws[i] = proj[b][c] / norm;
    }
}

// ---------------------------------------------------------------------------
// Kernel 2: bilinear gather + point-weighted sum + residual + LayerNorm.
// One wave32 per (b,q). Lane l owns channels c0 = 8*l .. 8*l+7 (head = l>>2),
// so every tap / residual / output access is a pair of 128-bit transactions.
// ---------------------------------------------------------------------------
__device__ __forceinline__
void tap8(const float* __restrict__ fb, int xi, int yi, int hh, int cb,
          int H, int W, float s[8])
{
    bool valid = (xi >= 0) & (xi < W) & (yi >= 0) & (yi < H);
    int xc = min(max(xi, 0), W - 1);
    int yc = min(max(yi, 0), H - 1);
    size_t off = ((size_t)(yc * W + xc) * NHD + hh) * DH + cb;
    f32x4 v0 = *(const f32x4*)(fb + off);
    f32x4 v1 = *(const f32x4*)(fb + off + 4);
    #pragma unroll
    for (int j = 0; j < 4; ++j) {
        s[j]     = valid ? v0[j] : 0.0f;
        s[4 + j] = valid ? v1[j] : 0.0f;
    }
}

__global__ void __launch_bounds__(256)
sample_ln_kernel(const float* __restrict__ feat,    // [bs, nq, NH, DH]
                 const float* __restrict__ ref2d,   // [bs, nq, 1, 2]
                 const float* __restrict__ query,   // [bs, 1, ED]
                 const float* __restrict__ ln_g,
                 const float* __restrict__ ln_b,
                 const float* __restrict__ off_ws,  // [bs,NH,NP,2]
                 const float* __restrict__ aw_ws,   // [bs,NH,NP]
                 const int*   __restrict__ hptr,
                 const int*   __restrict__ wptr,
                 float* __restrict__ out,           // [bs, nq, ED]
                 int bs, int nq)
{
    const int H = *hptr, W = *wptr;
    const int lane = threadIdx.x & 31;
    int wid = (int)blockIdx.x * ((int)blockDim.x >> 5) + ((int)threadIdx.x >> 5);
    wid = __builtin_amdgcn_readfirstlane(wid);   // wave-uniform -> SGPR/SMEM path
    if (wid >= bs * nq) return;
    const int b = wid / nq;

    const int hh = lane >> 2;        // head handled by this lane
    const int cb = (lane & 3) * 8;   // channel base within head
    const int c0 = hh * DH + cb;     // global channel base ( = 8*lane )

    const float rx = ref2d[(size_t)wid * 2 + 0];
    const float ry = ref2d[(size_t)wid * 2 + 1];
    const float* fb = feat + (size_t)b * nq * (NHD * DH);

    float racc[8];
    #pragma unroll
    for (int j = 0; j < 8; ++j) racc[j] = 0.0f;

    #pragma unroll
    for (int p = 0; p < NPT; ++p) {
        int oi = (b * NHD + hh) * NPT + p;
        float lx = rx + off_ws[2 * oi];
        float ly = ry + off_ws[2 * oi + 1];
        float aw = aw_ws[oi];
        // grid_sample(align_corners=False): x = loc_x*W - 0.5
        float x = lx * (float)W - 0.5f;
        float y = ly * (float)H - 0.5f;
        float x0f = floorf(x), y0f = floorf(y);
        int x0 = (int)x0f, y0 = (int)y0f;
        float wx = x - x0f, wy = y - y0f;

        float s00[8], s01[8], s10[8], s11[8];
        tap8(fb, x0,     y0,     hh, cb, H, W, s00);
        tap8(fb, x0 + 1, y0,     hh, cb, H, W, s01);
        tap8(fb, x0,     y0 + 1, hh, cb, H, W, s10);
        tap8(fb, x0 + 1, y0 + 1, hh, cb, H, W, s11);

        #pragma unroll
        for (int j = 0; j < 8; ++j) {
            float top = s00[j] + wx * (s01[j] - s00[j]);
            float bot = s10[j] + wx * (s11[j] - s10[j]);
            racc[j] = fmaf(aw, top + wy * (bot - top), racc[j]);
        }
    }

    // residual + LayerNorm over ED=256 (8 consecutive values/lane x 32 lanes)
    f32x4 q0 = *(const f32x4*)(query + b * EDIM + c0);
    f32x4 q1 = *(const f32x4*)(query + b * EDIM + c0 + 4);
    float s1 = 0.0f, s2 = 0.0f;
    #pragma unroll
    for (int j = 0; j < 4; ++j) {
        float r0 = racc[j] + q0[j];
        float r1 = racc[4 + j] + q1[j];
        racc[j] = r0; racc[4 + j] = r1;
        s1 += r0 + r1;
        s2 += r0 * r0 + r1 * r1;
    }
    #pragma unroll
    for (int m = 16; m >= 1; m >>= 1) {   // wave32 butterfly reduction
        s1 += __shfl_xor(s1, m, 32);
        s2 += __shfl_xor(s2, m, 32);
    }
    float mu  = s1 * (1.0f / EDIM);
    float var = s2 * (1.0f / EDIM) - mu * mu;
    float inv = rsqrtf(var + 1e-5f);

    f32x4 g0 = *(const f32x4*)(ln_g + c0);
    f32x4 g1 = *(const f32x4*)(ln_g + c0 + 4);
    f32x4 bb0 = *(const f32x4*)(ln_b + c0);
    f32x4 bb1 = *(const f32x4*)(ln_b + c0 + 4);

    f32x4 o0, o1;
    #pragma unroll
    for (int j = 0; j < 4; ++j) {
        o0[j] = (racc[j] - mu) * inv * g0[j] + bb0[j];
        o1[j] = (racc[4 + j] - mu) * inv * g1[j] + bb1[j];
    }
    float* op = out + (size_t)wid * EDIM + c0;
    *(f32x4*)(op)     = o0;
    *(f32x4*)(op + 4) = o1;
}

// ---------------------------------------------------------------------------
extern "C" void kernel_launch(void* const* d_in, const int* in_sizes, int n_in,
                              void* d_out, int out_size, void* d_ws, size_t ws_size,
                              hipStream_t stream)
{
    const float* query  = (const float*)d_in[0];
    const float* feat   = (const float*)d_in[1];
    const float* ref2d  = (const float*)d_in[2];
    const float* W_off  = (const float*)d_in[3];
    const float* b_off  = (const float*)d_in[4];
    const float* W_attn = (const float*)d_in[5];
    const float* b_attn = (const float*)d_in[6];
    const float* ln_g   = (const float*)d_in[7];
    const float* ln_b   = (const float*)d_in[8];
    const int*   hptr   = (const int*)d_in[9];
    const int*   wptr   = (const int*)d_in[10];

    const int bs = in_sizes[0] / EDIM;                 // query = [bs,1,ED]
    const int nq = in_sizes[1] / (bs * NHD * DH);      // single_feat = [bs,nq,NH,DH]

    float* off_ws = (float*)d_ws;                            // bs*NH*NP*2 floats
    float* aw_ws  = off_ws + (size_t)bs * NHD * NPT * 2;     // bs*NH*NP  floats

    hipLaunchKernelGGL(proj_wmma_kernel, dim3(1), dim3(384), 0, stream,
                       query, W_off, b_off, W_attn, b_attn, hptr, wptr,
                       off_ws, aw_ws, bs);

    const int waves_per_block = 256 / 32;
    const int total_waves = bs * nq;
    const int blocks = (total_waves + waves_per_block - 1) / waves_per_block;
    hipLaunchKernelGGL(sample_ln_kernel, dim3(blocks), dim3(256), 0, stream,
                       feat, ref2d, query, ln_g, ln_b, off_ws, aw_ws,
                       hptr, wptr, (float*)d_out, bs, nq);
}
